// MambaCrossAttnLayer_5557687681404
// MI455X (gfx1250) — matrix operand order repaired and checked by reading, not compiled, not run on hardware
//
#include <hip/hip_runtime.h>
#include <hip/hip_bf16.h>
#include <stdint.h>

// ---------------------------------------------------------------------------
// MambaCrossAttnLayer forward for MI455X (gfx1250, wave32, WMMA).
// All GEMMs (x @ W.T shape) run through v_wmma_f32_16x16x32_f16 with fp32
// accumulation. 64x32 wave tile (8 WMMA / K-step) for higher FLOP/byte.
// Scan / conv / softmax are VALU kernels. fp16 staging buffers live in d_ws.
// ---------------------------------------------------------------------------

typedef __attribute__((ext_vector_type(16))) _Float16 v16h;
typedef __attribute__((ext_vector_type(8)))  _Float16 v8h;
typedef __attribute__((ext_vector_type(8)))  float    v8f;

static constexpr int BB   = 8;
static constexpr int LP   = 1024;   // L_POSE
static constexpr int LT   = 128;    // L_TEXT
static constexpr int DM   = 512;    // D_MODEL
static constexpr int DI   = 1024;   // D_INNER
static constexpr int DS   = 16;     // D_STATE
static constexpr int DTR  = 32;     // DT_RANK
static constexpr int NH   = 8;
static constexpr int DH   = 64;

enum { ACT_NONE = 0, ACT_SILU = 1, ACT_GELU = 2, ACT_SOFTPLUS = 3 };

__device__ __forceinline__ v16h cat8(v8h lo, v8h hi) {
  v16h r;
#pragma unroll
  for (int i = 0; i < 8; ++i) { r[i] = lo[i]; r[i + 8] = hi[i]; }
  return r;
}

__device__ __forceinline__ float silu_f(float x) { return x / (1.f + __expf(-x)); }

// ---------------------------------------------------------------------------
// Generic f32 -> f16 conversion
// ---------------------------------------------------------------------------
__global__ void cvt_f32_f16_kernel(const float* __restrict__ src,
                                   _Float16* __restrict__ dst, int n) {
  int i = blockIdx.x * 256 + threadIdx.x;
  if (i < n) dst[i] = (_Float16)src[i];
}

// ---------------------------------------------------------------------------
// LayerNorm (cols == 512), one row per 256-thread block, f16 output
// ---------------------------------------------------------------------------
__global__ void __launch_bounds__(256) layernorm_kernel(
    const float* __restrict__ x, const float* __restrict__ g,
    const float* __restrict__ b, _Float16* __restrict__ out) {
  const int row  = blockIdx.x;
  const int tid  = threadIdx.x;
  const int wave = tid >> 5, lane = tid & 31;
  __shared__ float red[8];

  const float* xr = x + (size_t)row * DM;
  float v0 = xr[tid], v1 = xr[tid + 256];

  float s = v0 + v1;
#pragma unroll
  for (int o = 16; o > 0; o >>= 1) s += __shfl_xor(s, o, 32);
  if (lane == 0) red[wave] = s;
  __syncthreads();
  float tot = 0.f;
#pragma unroll
  for (int i = 0; i < 8; ++i) tot += red[i];
  const float mean = tot * (1.f / (float)DM);
  __syncthreads();

  float d0 = v0 - mean, d1 = v1 - mean;
  s = d0 * d0 + d1 * d1;
#pragma unroll
  for (int o = 16; o > 0; o >>= 1) s += __shfl_xor(s, o, 32);
  if (lane == 0) red[wave] = s;
  __syncthreads();
  tot = 0.f;
#pragma unroll
  for (int i = 0; i < 8; ++i) tot += red[i];
  const float inv = rsqrtf(tot * (1.f / (float)DM) + 1e-5f);

  _Float16* orow = out + (size_t)row * DM;
  orow[tid]       = (_Float16)(d0 * inv * g[tid] + b[tid]);
  orow[tid + 256] = (_Float16)(d1 * inv * g[tid + 256] + b[tid + 256]);
}

// ---------------------------------------------------------------------------
// WMMA GEMM:  C[M,N] = epilogue( A16[M,K] x B16[N,K]^T + bias )
// Wave tile 64x32 (4x2 fragments, 8 WMMA per K=32 step), block tile 256x64
// (8 waves as 4x2).
// A fragment: lane = M row, K split {g*8 .. g*8+7} U {16+g*8 .. 16+g*8+7}
// B fragment: lane = N row, K split {g*8 .. g*8+7} U {16+g*8 .. 16+g*8+7}
// C fragment: lane -> col n0+(lane&15), VGPR v -> row m0 + v + 8*(lane>>4)
// ---------------------------------------------------------------------------
template <int ACT, bool OUT16, bool RESID, bool MASK>
__global__ void __launch_bounds__(256) gemm_wmma_kernel(
    const _Float16* __restrict__ A, const _Float16* __restrict__ B,
    const float* __restrict__ bias, void* __restrict__ C,
    const float* __restrict__ resid, const uint8_t* __restrict__ mask,
    int M, int N, int K, int lda, int ldb, int ldc) {
  const int tid  = threadIdx.x;
  const int wave = tid >> 5;
  const int lane = tid & 31;
  const int r = lane & 15;
  const int g = lane >> 4;
  const int m0 = blockIdx.x * 256 + (wave >> 1) * 64;
  const int n0 = blockIdx.y * 64 + (wave & 1) * 32;
  if (m0 >= M || n0 >= N) return;

  v8f c[4][2] = {};

  const _Float16* pa = A + (size_t)(m0 + r) * lda + g * 8;
  const _Float16* pb = B + (size_t)(n0 + r) * ldb + g * 8;
  const size_t a16 = (size_t)16 * lda;
  const size_t b16 = (size_t)16 * ldb;

  for (int k0 = 0; k0 < K; k0 += 32) {
    __builtin_prefetch((const void*)(pa + k0 + 32), 0, 1);
    v16h a[4], bfr[2];
#pragma unroll
    for (int i = 0; i < 4; ++i) {
      const _Float16* p = pa + (size_t)i * a16 + k0;
      a[i] = cat8(*(const v8h*)p, *(const v8h*)(p + 16));
    }
#pragma unroll
    for (int j = 0; j < 2; ++j) {
      const _Float16* p = pb + (size_t)j * b16 + k0;
      bfr[j] = cat8(*(const v8h*)p, *(const v8h*)(p + 16));
    }
#pragma unroll
    for (int i = 0; i < 4; ++i) {
#pragma unroll
      for (int j = 0; j < 2; ++j) {
        c[i][j] = __builtin_amdgcn_wmma_f32_16x16x32_f16(
            false, a[i], false, bfr[j], (short)0, c[i][j], false, false);
      }
    }
  }

#pragma unroll
  for (int i = 0; i < 4; ++i) {
#pragma unroll
    for (int j = 0; j < 2; ++j) {
      const int nn = n0 + j * 16 + r;
      const float bv = bias ? bias[nn] : 0.f;
#pragma unroll
      for (int v = 0; v < 8; ++v) {
        const int mm = m0 + i * 16 + g * 8 + v;
        float x = c[i][j][v] + bv;
        if (ACT == ACT_SILU)      x = silu_f(x);
        else if (ACT == ACT_GELU) x = 0.5f * x * (1.f + erff(x * 0.70710678118f));
        else if (ACT == ACT_SOFTPLUS) x = (x > 20.f) ? x : log1pf(__expf(x));
        if (MASK) { if (mask[mm]) x = 0.f; }
        const size_t idx = (size_t)mm * ldc + nn;
        if (RESID) x += resid[idx];
        if (OUT16) ((_Float16*)C)[idx] = (_Float16)x;
        else       ((float*)C)[idx] = x;
      }
    }
  }
}

// ---------------------------------------------------------------------------
// Depthwise causal conv (D_CONV=4) + bias + SiLU on u (first half of xz)
// ---------------------------------------------------------------------------
__global__ void __launch_bounds__(256) conv_silu_kernel(
    const _Float16* __restrict__ xz, const float* __restrict__ w,
    const float* __restrict__ bias, float* __restrict__ u32,
    _Float16* __restrict__ u16) {
  const size_t i = (size_t)blockIdx.x * 256 + threadIdx.x;  // (b*L+l)*DI + d
  const int d = (int)(i & (DI - 1));
  const size_t bl = i >> 10;              // b*L + l  (DI == 1024)
  const int l = (int)(bl & (LP - 1));     // L == 1024
  const size_t brow = bl - l;             // b*L
  const float* wd = w + d * 4;
  float acc = bias[d];
#pragma unroll
  for (int j = 0; j < 4; ++j) {
    const int t = l - 3 + j;
    if (t >= 0) acc += wd[j] * (float)xz[(brow + t) * (size_t)(2 * DI) + d];
  }
  const float s = silu_f(acc);
  u32[i] = s;
  u16[i] = (_Float16)s;
}

// ---------------------------------------------------------------------------
// Selective scan: one thread per (b, d) channel, sequential over L.
// y = (scan + u*D) * silu(z); z taken from second half of xz.
// ---------------------------------------------------------------------------
__global__ void __launch_bounds__(256) scan_kernel(
    const float* __restrict__ u32, const float* __restrict__ dtf,
    const float* __restrict__ xdbl, const _Float16* __restrict__ xz,
    const float* __restrict__ A_log, const float* __restrict__ Dskip,
    _Float16* __restrict__ y16) {
  const int idx = blockIdx.x * 256 + threadIdx.x;  // 0 .. B*DI-1
  const int d = idx & (DI - 1);
  const int b = idx >> 10;

  float Av[DS];
#pragma unroll
  for (int s = 0; s < DS; ++s) Av[s] = -__expf(A_log[d * DS + s]);
  const float Dd = Dskip[d];

  float h[DS];
#pragma unroll
  for (int s = 0; s < DS; ++s) h[s] = 0.f;

  for (int t = 0; t < LP; ++t) {
    const size_t rl = (size_t)b * LP + t;
    const float uv  = u32[rl * DI + d];
    const float dtv = dtf[rl * DI + d];
    const float du  = dtv * uv;
    const float* xr = xdbl + rl * (DTR + 2 * DS);
    float y = 0.f;
#pragma unroll
    for (int s = 0; s < DS; ++s) {
      const float dA = __expf(dtv * Av[s]);
      h[s] = h[s] * dA + du * xr[DTR + s];
      y += h[s] * xr[DTR + DS + s];
    }
    y += uv * Dd;
    const float zv = (float)xz[rl * (size_t)(2 * DI) + DI + d];
    y *= silu_f(zv);
    y16[rl * DI + d] = (_Float16)y;
  }
}

// ---------------------------------------------------------------------------
// Cross attention (VALU): one wave per (b, h, q) row; keys = 128, dh = 64.
// Wave shuffle softmax; per-wave LDS buffer for probabilities.
// ---------------------------------------------------------------------------
__global__ void __launch_bounds__(256) attn_kernel(
    const float* __restrict__ Q, const float* __restrict__ Kf,
    const float* __restrict__ Vf, const uint8_t* __restrict__ tmask,
    _Float16* __restrict__ o16) {
  __shared__ float pbuf[8][LT];
  const int tid  = threadIdx.x;
  const int wave = tid >> 5, lane = tid & 31;
  const int row = blockIdx.x * 8 + wave;          // b*NH*LP + h*LP + q
  const int b = row >> 13;                        // NH*LP = 8192
  const int rem = row & 8191;
  const int h = rem >> 10;
  const int q = rem & (LP - 1);

  const float* qp = Q + ((size_t)b * LP + q) * DM + h * DH;

  float sc[4];
#pragma unroll
  for (int kk = 0; kk < 4; ++kk) {
    const int k = kk * 32 + lane;
    const float* kp = Kf + ((size_t)b * LT + k) * DM + h * DH;
    float s = 0.f;
#pragma unroll
    for (int j = 0; j < DH; ++j) s += qp[j] * kp[j];
    s *= 0.125f;  // 1/sqrt(64)
    if (tmask[b * LT + k]) s = -1e9f;
    sc[kk] = s;
  }
  float m = fmaxf(fmaxf(sc[0], sc[1]), fmaxf(sc[2], sc[3]));
#pragma unroll
  for (int o = 16; o > 0; o >>= 1) m = fmaxf(m, __shfl_xor(m, o, 32));
  float p[4], sum = 0.f;
#pragma unroll
  for (int kk = 0; kk < 4; ++kk) { p[kk] = __expf(sc[kk] - m); sum += p[kk]; }
#pragma unroll
  for (int o = 16; o > 0; o >>= 1) sum += __shfl_xor(sum, o, 32);
  const float inv = 1.f / sum;
#pragma unroll
  for (int kk = 0; kk < 4; ++kk) pbuf[wave][kk * 32 + lane] = p[kk] * inv;
  __syncthreads();

  const int j0 = lane * 2;
  float o0 = 0.f, o1 = 0.f;
  for (int k = 0; k < LT; ++k) {
    const float pk = pbuf[wave][k];
    const float* vp = Vf + ((size_t)b * LT + k) * DM + h * DH;
    o0 += pk * vp[j0];
    o1 += pk * vp[j0 + 1];
  }
  _Float16* op = o16 + ((size_t)b * LP + q) * DM + h * DH;
  op[j0]     = (_Float16)o0;
  op[j0 + 1] = (_Float16)o1;
}

// ---------------------------------------------------------------------------
// Host launcher
// ---------------------------------------------------------------------------
extern "C" void kernel_launch(void* const* d_in, const int* in_sizes, int n_in,
                              void* d_out, int out_size, void* d_ws, size_t ws_size,
                              hipStream_t stream) {
  (void)in_sizes; (void)n_in; (void)out_size; (void)ws_size;
  const float* z_t        = (const float*)d_in[0];
  const float* text       = (const float*)d_in[1];
  const uint8_t* text_msk = (const uint8_t*)d_in[2];
  const uint8_t* pose_msk = (const uint8_t*)d_in[3];
  const float* g_mamba = (const float*)d_in[4];
  const float* b_mamba = (const float*)d_in[5];
  const float* in_proj_w = (const float*)d_in[6];
  const float* conv_w    = (const float*)d_in[7];
  const float* conv_b    = (const float*)d_in[8];
  const float* x_proj_w  = (const float*)d_in[9];
  const float* dt_proj_w = (const float*)d_in[10];
  const float* dt_proj_b = (const float*)d_in[11];
  const float* A_log     = (const float*)d_in[12];
  const float* D_skip    = (const float*)d_in[13];
  const float* out_proj_w = (const float*)d_in[14];
  const float* g_cross = (const float*)d_in[15];
  const float* b_cross = (const float*)d_in[16];
  const float* attn_in_w  = (const float*)d_in[17];
  const float* attn_in_b  = (const float*)d_in[18];
  const float* attn_out_w = (const float*)d_in[19];
  const float* attn_out_b = (const float*)d_in[20];
  const float* g_ffn = (const float*)d_in[21];
  const float* b_ffn = (const float*)d_in[22];
  const float* w1 = (const float*)d_in[23];
  const float* b1 = (const float*)d_in[24];
  const float* w2 = (const float*)d_in[25];
  const float* b2 = (const float*)d_in[26];

  const int MR = BB * LP;   // 8192 pose tokens
  const int MT = BB * LT;   // 1024 text tokens

  // ---- workspace bump allocator ----
  char* ws = (char*)d_ws;
  size_t off = 0;
  auto alloc = [&](size_t bytes) -> void* {
    void* p = ws + off;
    off = (off + bytes + 255) & ~(size_t)255;
    return p;
  };
  _Float16* ln1_16   = (_Float16*)alloc((size_t)MR * DM * 2);
  _Float16* inpr16   = (_Float16*)alloc((size_t)2 * DI * DM * 2);
  _Float16* xproj16  = (_Float16*)alloc((size_t)(DTR + 2 * DS) * DI * 2);
  _Float16* dtpr16   = (_Float16*)alloc((size_t)DI * DTR * 2);
  _Float16* outpr16  = (_Float16*)alloc((size_t)DM * DI * 2);
  _Float16* ain16    = (_Float16*)alloc((size_t)3 * DM * DM * 2);
  _Float16* aout16   = (_Float16*)alloc((size_t)DM * DM * 2);
  _Float16* w1_16    = (_Float16*)alloc((size_t)4 * DM * DM * 2);
  _Float16* w2_16    = (_Float16*)alloc((size_t)DM * 4 * DM * 2);
  _Float16* text16   = (_Float16*)alloc((size_t)MT * DM * 2);
  _Float16* xz16     = (_Float16*)alloc((size_t)MR * 2 * DI * 2);
  float*    u32      = (float*)alloc((size_t)MR * DI * 4);
  _Float16* u16      = (_Float16*)alloc((size_t)MR * DI * 2);
  float*    xdbl32   = (float*)alloc((size_t)MR * (DTR + 2 * DS) * 4);
  _Float16* xdbl16   = (_Float16*)alloc((size_t)MR * (DTR + 2 * DS) * 2);
  float*    dt32     = (float*)alloc((size_t)MR * DI * 4);
  _Float16* y16      = (_Float16*)alloc((size_t)MR * DI * 2);
  float*    zres     = (float*)alloc((size_t)MR * DM * 4);
  _Float16* ln2_16   = (_Float16*)alloc((size_t)MR * DM * 2);
  float*    Qf       = (float*)alloc((size_t)MR * DM * 4);
  float*    Kfb      = (float*)alloc((size_t)MT * DM * 4);
  float*    Vfb      = (float*)alloc((size_t)MT * DM * 4);
  _Float16* o16      = (_Float16*)alloc((size_t)MR * DM * 2);
  float*    z2       = (float*)alloc((size_t)MR * DM * 4);
  _Float16* ln3_16   = (_Float16*)alloc((size_t)MR * DM * 2);
  _Float16* h1_16    = (_Float16*)alloc((size_t)MR * 4 * DM * 2);
  float*    outf     = (float*)d_out;

  auto cvt = [&](const float* s, _Float16* d, int n) {
    cvt_f32_f16_kernel<<<(n + 255) / 256, 256, 0, stream>>>(s, d, n);
  };

  // ---- weight / activation f16 staging ----
  cvt(in_proj_w, inpr16, 2 * DI * DM);
  cvt(x_proj_w, xproj16, (DTR + 2 * DS) * DI);
  cvt(dt_proj_w, dtpr16, DI * DTR);
  cvt(out_proj_w, outpr16, DM * DI);
  cvt(attn_in_w, ain16, 3 * DM * DM);
  cvt(attn_out_w, aout16, DM * DM);
  cvt(w1, w1_16, 4 * DM * DM);
  cvt(w2, w2_16, DM * 4 * DM);
  cvt(text, text16, MT * DM);

  const dim3 blk(256);
  auto gg = [](int M, int N) { return dim3(M / 256, N / 64, 1); };

  // ---- Mamba block ----
  layernorm_kernel<<<MR, blk, 0, stream>>>(z_t, g_mamba, b_mamba, ln1_16);

  // xz = ln1 @ in_proj_w.T : (8192 x 2048), K=512, f16 out
  gemm_wmma_kernel<ACT_NONE, true, false, false><<<gg(MR, 2 * DI), blk, 0, stream>>>(
      ln1_16, inpr16, nullptr, xz16, nullptr, nullptr, MR, 2 * DI, DM, DM, DM, 2 * DI);

  conv_silu_kernel<<<(MR * DI) / 256, blk, 0, stream>>>(xz16, conv_w, conv_b, u32, u16);

  // x_dbl = u @ x_proj_w.T : (8192 x 64), K=1024, f32 out
  gemm_wmma_kernel<ACT_NONE, false, false, false><<<gg(MR, DTR + 2 * DS), blk, 0, stream>>>(
      u16, xproj16, nullptr, xdbl32, nullptr, nullptr, MR, DTR + 2 * DS, DI, DI, DI, DTR + 2 * DS);
  cvt(xdbl32, xdbl16, MR * (DTR + 2 * DS));

  // dt = softplus(x_dbl[:, :32] @ dt_proj_w.T + b) : (8192 x 1024), K=32
  gemm_wmma_kernel<ACT_SOFTPLUS, false, false, false><<<gg(MR, DI), blk, 0, stream>>>(
      xdbl16, dtpr16, dt_proj_b, dt32, nullptr, nullptr, MR, DI, DTR, DTR + 2 * DS, DTR, DI);

  scan_kernel<<<(BB * DI) / 256, blk, 0, stream>>>(u32, dt32, xdbl32, xz16, A_log, D_skip, y16);

  // z = z_t + where(pose_mask, 0, y @ out_proj_w.T) : (8192 x 512), K=1024
  gemm_wmma_kernel<ACT_NONE, false, true, true><<<gg(MR, DM), blk, 0, stream>>>(
      y16, outpr16, nullptr, zres, z_t, pose_msk, MR, DM, DI, DI, DI, DM);

  // ---- Cross attention ----
  layernorm_kernel<<<MR, blk, 0, stream>>>(zres, g_cross, b_cross, ln2_16);

  gemm_wmma_kernel<ACT_NONE, false, false, false><<<gg(MR, DM), blk, 0, stream>>>(
      ln2_16, ain16, attn_in_b, Qf, nullptr, nullptr, MR, DM, DM, DM, DM, DM);
  gemm_wmma_kernel<ACT_NONE, false, false, false><<<gg(MT, DM), blk, 0, stream>>>(
      text16, ain16 + (size_t)DM * DM, attn_in_b + DM, Kfb, nullptr, nullptr, MT, DM, DM, DM, DM, DM);
  gemm_wmma_kernel<ACT_NONE, false, false, false><<<gg(MT, DM), blk, 0, stream>>>(
      text16, ain16 + (size_t)2 * DM * DM, attn_in_b + 2 * DM, Vfb, nullptr, nullptr, MT, DM, DM, DM, DM, DM);

  attn_kernel<<<(BB * NH * LP) / 8, blk, 0, stream>>>(Qf, Kfb, Vfb, text_msk, o16);

  // z2 = z + o @ attn_out_w.T + b : (8192 x 512), K=512
  gemm_wmma_kernel<ACT_NONE, false, true, false><<<gg(MR, DM), blk, 0, stream>>>(
      o16, aout16, attn_out_b, z2, zres, nullptr, MR, DM, DM, DM, DM, DM);

  // ---- FFN ----
  layernorm_kernel<<<MR, blk, 0, stream>>>(z2, g_ffn, b_ffn, ln3_16);

  gemm_wmma_kernel<ACT_GELU, true, false, false><<<gg(MR, 4 * DM), blk, 0, stream>>>(
      ln3_16, w1_16, b1, h1_16, nullptr, nullptr, MR, 4 * DM, DM, DM, DM, 4 * DM);

  gemm_wmma_kernel<ACT_NONE, false, true, false><<<gg(MR, DM), blk, 0, stream>>>(
      h1_16, w2_16, b2, outf, z2, nullptr, MR, DM, 4 * DM, 4 * DM, 4 * DM, DM);
}
